// HuMAn_21638045237767
// MI455X (gfx1250) — matrix operand order repaired and checked by reading, not compiled, not run on hardware
//
#include <hip/hip_runtime.h>
#include <hip/hip_bf16.h>

typedef _Float16 h16;
typedef __attribute__((ext_vector_type(8)))  _Float16 v8h;
typedef __attribute__((ext_vector_type(16))) _Float16 v16h;
typedef __attribute__((ext_vector_type(8)))  float    v8f;

#define WMMA_F16(a, b, c) \
  __builtin_amdgcn_wmma_f32_16x16x32_f16(false, (a), false, (b), (short)0, (c), false, false)

// ---- problem constants ----
#define BB 16
#define TT 1024
#define MM (BB * TT)      // 16384 rows
#define HH 1024           // LSTM units
#define N4 4096           // 4*H
#define KX 96             // padded 73 -> 96
#define FD 1056           // padded feature dim

// ============================================================================
// WMMA fragment loaders (wave32, contiguous k order k = 16*(lane/16) + s in A and B) -- vectorized
// ============================================================================
__device__ inline v16h load_a_frag(const h16* __restrict__ src, int ld, int kbase) {
  int lane = threadIdx.x & 31;
  int row  = lane & 15;
  int khi  = (lane & 16) ? 16 : 0;
  const h16* p = src + (size_t)row * ld + kbase + khi;
  v8h lo = *(const v8h*)p;
  v8h hi = *(const v8h*)(p + 8);
  return __builtin_shufflevector(lo, hi, 0,1,2,3,4,5,6,7,8,9,10,11,12,13,14,15);
}

// B 32x16 f16 from fragment-linear (swizzled) storage: one 32-byte load per lane.
__device__ inline v16h load_b_swz(const h16* __restrict__ src, int ntiles, int kt, int nt) {
  int lane = threadIdx.x & 31;
  size_t base = (((size_t)(kt * ntiles + nt)) * 32 + lane) * 16;
  return *(const v16h*)(src + base);
}

__device__ inline float sigm(float x) { return 1.0f / (1.0f + __expf(-x)); }

// ============================================================================
// elementwise prep / conversion kernels
// ============================================================================
__global__ void prep_x(const float* __restrict__ pose, const float* __restrict__ sel,
                       const float* __restrict__ elapsed, const float* __restrict__ mean,
                       const float* __restrict__ var, h16* __restrict__ xpad) {
  int idx = blockIdx.x * blockDim.x + threadIdx.x;
  if (idx >= MM * KX) return;
  int m = idx / KX, k = idx % KX;
  float v;
  if (k < 72)       v = (pose[(size_t)m * 72 + k] - mean[k]) / fmaxf(sqrtf(var[k]), 1e-7f)
                        * sel[(size_t)m * 72 + k];
  else if (k == 72) v = elapsed[m];
  else              v = 0.0f;
  xpad[idx] = (h16)v;
}

__global__ void f32_to_f16_pad(const float* __restrict__ src, h16* __restrict__ dst,
                               int rows, int cols, int padRows) {
  int idx = blockIdx.x * blockDim.x + threadIdx.x;
  if (idx >= padRows * cols) return;
  int r = idx / cols, c = idx % cols;
  dst[idx] = (r < rows) ? (h16)src[(size_t)r * cols + c] : (h16)0.0f;
}

// f32 -> f16 into fragment-linear B layout (zero K padding)
__global__ void f32_to_f16_swz(const float* __restrict__ src, h16* __restrict__ dst,
                               int rows, int cols, int padK) {
  int idx = blockIdx.x * blockDim.x + threadIdx.x;
  if (idx >= padK * cols) return;
  int k = idx / cols, n = idx % cols;
  float v = (k < rows) ? src[(size_t)k * cols + n] : 0.0f;
  int kt = k >> 5, kg = (k >> 4) & 1, h = k & 15;
  int nt = n >> 4, nl = n & 15;
  int ntiles = cols >> 4;
  size_t off = (((size_t)(kt * ntiles + nt)) * 32 + kg * 16 + nl) * 16 + h;
  dst[off] = (h16)v;
}

__global__ void feat_base(const float* __restrict__ elapsed, const float* __restrict__ horizon,
                          h16* __restrict__ feat) {
  int m = blockIdx.x * blockDim.x + threadIdx.x;
  if (m >= MM) return;
  feat[(size_t)m * FD + 0] = (h16)elapsed[m];
  feat[(size_t)m * FD + 1] = (h16)horizon[m];
}

// ============================================================================
// xW = Xpad[16384,96] @ Wp_swz[96,4096] + bias -> f16 [16384,4096]
// ============================================================================
__global__ __launch_bounds__(128) void xw_gemm(const h16* __restrict__ xpad,
                                               const h16* __restrict__ wp,
                                               const float* __restrict__ bias,
                                               h16* __restrict__ xw) {
  int wave  = threadIdx.x >> 5;
  int mbase = blockIdx.x * 16;
  int nbase = (blockIdx.y * 4 + wave) * 16;
  v8f acc = {};
#pragma unroll
  for (int kc = 0; kc < KX; kc += 32) {
    v16h a = load_a_frag(xpad + (size_t)mbase * KX, KX, kc);
    v16h b = load_b_swz(wp, N4 / 16, kc >> 5, nbase >> 4);
    acc = WMMA_F16(a, b, acc);
  }
  int lane = threadIdx.x & 31;
  int col  = nbase + (lane & 15);
  int rb   = (lane >> 4) * 8;
#pragma unroll
  for (int r = 0; r < 8; ++r)
    xw[(size_t)(mbase + rb + r) * N4 + col] = (h16)(acc[r] + bias[col]);
}

// ============================================================================
// cooperative LSTM
// ============================================================================
__device__ inline void grid_barrier(unsigned int* cnt, unsigned int target) {
  __syncthreads();
  __threadfence();
  if (threadIdx.x == 0) {
    atomicAdd(cnt, 1u);
    while (__hip_atomic_load(cnt, __ATOMIC_ACQUIRE, __HIP_MEMORY_SCOPE_AGENT) < target)
      __builtin_amdgcn_s_sleep(1);
  }
  __threadfence();
  __syncthreads();
}

__device__ inline void add_z(float* zlds, v8f acc, int colbase) {
  int lane = threadIdx.x & 31;
  int cc   = colbase + (lane & 15);
  int rb   = (lane >> 4) * 8;
#pragma unroll
  for (int r = 0; r < 8; ++r) zlds[(rb + r) * 128 + cc] += acc[r];
}

__global__ __launch_bounds__(128) void lstm_kernel(const h16* __restrict__ xw,
                                                   const h16* __restrict__ u16,
                                                   const float* __restrict__ bias,
                                                   h16* __restrict__ hbuf,
                                                   h16* __restrict__ feat,
                                                   unsigned int* __restrict__ cnt) {
  extern __shared__ char smem[];
  h16*   ulds = (h16*)smem;                                          // 256 KB (swizzled)
  h16*   hlds = (h16*)(smem + (size_t)HH * 128 * 2);                 // 32 KB
  float* zlds = (float*)(smem + (size_t)HH * 128 * 2 + BB * HH * 2); // 8 KB
  float* clds = zlds + BB * 128;                                     // 2 KB
  float* blds = clds + BB * 32;                                      // 512 B

  int tid = threadIdx.x, wg = blockIdx.x, hbase = wg * 32;

  // pin this block's U slice into LDS in fragment-linear layout (local cols = 128)
  for (int idx = tid; idx < HH * 128; idx += 128) {
    int k = idx >> 7, c = idx & 127, g = c >> 5;
    int kt = k >> 5, kg = (k >> 4) & 1, h = k & 15;
    int nt = c >> 4, nl = c & 15;
    size_t off = (((size_t)(kt * 8 + nt)) * 32 + kg * 16 + nl) * 16 + h;
    ulds[off] = u16[(size_t)k * N4 + g * HH + hbase + (c & 31)];
  }
  if (tid < 128) blds[tid] = bias[(tid >> 5) * HH + hbase + (tid & 31)];
  for (int idx = tid; idx < BB * 32; idx += 128) clds[idx] = 0.0f;
  for (int idx = tid; idx < BB * 32; idx += 128) {       // h0 = 0 (parity 0)
    int b = idx >> 5, n = idx & 31;
    hbuf[(size_t)b * HH + hbase + n] = (h16)0.0f;
  }
  grid_barrier(cnt, 32u);

  int g = tid >> 5;   // wave = gate
  for (int t = 0; t < TT; ++t) {
    const h16* hcur = hbuf + (size_t)(t & 1) * (BB * HH);
    h16*       hnxt = hbuf + (size_t)((t + 1) & 1) * (BB * HH);

    for (int i = tid; i < BB * HH / 8; i += 128)          // h broadcast -> LDS, 16B
      ((uint4*)hlds)[i] = ((const uint4*)hcur)[i];
    {                                                     // z := xW + bias (32B loads)
      int b = tid >> 3, gg = (tid >> 1) & 3, part = tid & 1;
      size_t m = (size_t)b * TT + t;
      v16h xv = *(const v16h*)(xw + m * N4 + gg * HH + hbase + part * 16);
      int cb = gg * 32 + part * 16;
#pragma unroll
      for (int i = 0; i < 16; ++i)
        zlds[b * 128 + cb + i] = (float)xv[i] + blds[cb + i];
    }
    __syncthreads();

    v8f acc0 = {}, acc1 = {};
    for (int kc = 0; kc < HH; kc += 32) {
      int kt = kc >> 5;
      v16h a  = load_a_frag(hlds, HH, kc);
      v16h b0 = load_b_swz(ulds, 8, kt, g * 2);
      v16h b1 = load_b_swz(ulds, 8, kt, g * 2 + 1);
      acc0 = WMMA_F16(a, b0, acc0);
      acc1 = WMMA_F16(a, b1, acc1);
    }
    add_z(zlds, acc0, g * 32);
    add_z(zlds, acc1, g * 32 + 16);
    __syncthreads();

    for (int idx = tid; idx < BB * 32; idx += 128) {
      int b = idx >> 5, n = idx & 31;
      float zi = zlds[b * 128 + n];
      float zf = zlds[b * 128 + 32 + n];
      float zg = zlds[b * 128 + 64 + n];
      float zo = zlds[b * 128 + 96 + n];
      float cn = sigm(zf) * clds[idx] + sigm(zi) * tanhf(zg);
      clds[idx] = cn;
      float hn = sigm(zo) * tanhf(cn);
      hnxt[(size_t)b * HH + hbase + n] = (h16)hn;
      feat[(size_t)(b * TT + t) * FD + 2 + hbase + n] = (h16)hn;
    }
    if (t + 1 < TT) {  // global_prefetch_b8 of next step's xW slice
      const h16* p = xw + (size_t)((tid & 15) * TT + t + 1) * N4 + (tid >> 4) * 512;
      __builtin_prefetch((const void*)p, 0, 1);
    }
    grid_barrier(cnt, 32u * (unsigned)(t + 2));
  }
}

// ============================================================================
// per-joint MLPs -- layers 0..2 on WMMA, layer 3 (N=3) scalar
// ============================================================================
struct Parents { int p[8]; int np; };

__global__ void fill_parents(const float* __restrict__ deltas, h16* __restrict__ feat,
                             Parents par) {
  int m = blockIdx.x * blockDim.x + threadIdx.x;
  if (m >= MM) return;
  for (int c = 0; c < 30; ++c) {
    float v = 0.0f;
    if (c < 3 * par.np) v = deltas[(size_t)m * 72 + 3 * par.p[c / 3] + (c % 3)];
    feat[(size_t)m * FD + 1026 + c] = (h16)v;
  }
}

__global__ __launch_bounds__(32) void joint_mlp(const h16* __restrict__ feat,
                                                const h16* __restrict__ w0p,
                                                const float* __restrict__ b0,
                                                const h16* __restrict__ w1p,
                                                const float* __restrict__ b1,
                                                const h16* __restrict__ w2p,
                                                const float* __restrict__ b2,
                                                const float* __restrict__ w3,
                                                const float* __restrict__ b3,
                                                const float* __restrict__ sel,
                                                float* __restrict__ deltas,
                                                int j3, int kpad) {
  __shared__ h16 h0[16 * 64];
  __shared__ h16 h1[16 * 32];
  __shared__ h16 h2[16 * 32];
  int mbase = blockIdx.x * 16;
  int lane  = threadIdx.x & 31;
  int col = lane & 15, rb = (lane >> 4) * 8;

  v8f z = {};
  // ---- layer 0: [16,1056] @ [1056,64], WMMA ----
  v8f acc[4] = {z, z, z, z};
  for (int kc = 0; kc < kpad; kc += 32) {
    v16h a = load_a_frag(feat + (size_t)mbase * FD, FD, kc);
#pragma unroll
    for (int nt = 0; nt < 4; ++nt)
      acc[nt] = WMMA_F16(a, load_b_swz(w0p, 4, kc >> 5, nt), acc[nt]);
  }
#pragma unroll
  for (int nt = 0; nt < 4; ++nt)
#pragma unroll
    for (int r = 0; r < 8; ++r)
      h0[(rb + r) * 64 + nt * 16 + col] = (h16)tanhf(acc[nt][r] + b0[nt * 16 + col]);
  __syncthreads();

  // ---- layer 1: [16,64] @ [64,32], WMMA ----
  v8f a1[2] = {z, z};
#pragma unroll
  for (int kc = 0; kc < 64; kc += 32) {
    v16h a = load_a_frag(h0, 64, kc);
#pragma unroll
    for (int nt = 0; nt < 2; ++nt)
      a1[nt] = WMMA_F16(a, load_b_swz(w1p, 2, kc >> 5, nt), a1[nt]);
  }
#pragma unroll
  for (int nt = 0; nt < 2; ++nt)
#pragma unroll
    for (int r = 0; r < 8; ++r)
      h1[(rb + r) * 32 + nt * 16 + col] = (h16)tanhf(a1[nt][r] + b1[nt * 16 + col]);
  __syncthreads();

  // ---- layer 2: [16,32] @ [32,32], WMMA ----
  v8f a2[2] = {z, z};
  {
    v16h a = load_a_frag(h1, 32, 0);
#pragma unroll
    for (int nt = 0; nt < 2; ++nt)
      a2[nt] = WMMA_F16(a, load_b_swz(w2p, 2, 0, nt), a2[nt]);
  }
#pragma unroll
  for (int nt = 0; nt < 2; ++nt)
#pragma unroll
    for (int r = 0; r < 8; ++r)
      h2[(rb + r) * 32 + nt * 16 + col] = (h16)tanhf(a2[nt][r] + b2[nt * 16 + col]);
  __syncthreads();

  // ---- layer 3: [16,32] @ [32,3], scalar, * selection ----
  for (int idx = lane; idx < 48; idx += 32) {
    int r = idx / 3, c = idx % 3;
    float s = b3[c];
    for (int k = 0; k < 32; ++k) s += (float)h2[r * 32 + k] * w3[k * 3 + c];
    int m = mbase + r;
    deltas[(size_t)m * 72 + j3 + c] = sel[(size_t)m * 72 + j3 + c] * s;
  }
}

__global__ void final_pred(const float* __restrict__ pose, const float* __restrict__ sel,
                           const float* __restrict__ deltas, float* __restrict__ out) {
  int idx = blockIdx.x * blockDim.x + threadIdx.x;
  if (idx >= MM * 72) return;
  out[idx] = pose[idx] * sel[idx] + deltas[idx];
}

// ============================================================================
// host launcher
// ============================================================================
static const int PN[24] = {0,1,1,1,2,2,2,3,3,3,4,4,4,4,4,5,5,5,6,6,7,7,8,8};
static const int PL[24][8] = {
  {0},{0},{0},{0},{0,1},{0,2},{0,3},{0,1,4},{0,2,5},{0,3,6},
  {0,1,4,7},{0,2,5,8},{0,3,6,9},{0,3,6,9},{0,3,6,9},
  {0,3,6,9,12},{0,3,6,9,13},{0,3,6,9,14},
  {0,3,6,9,13,16},{0,3,6,9,14,17},
  {0,3,6,9,13,16,18},{0,3,6,9,14,17,19},
  {0,3,6,9,13,16,18,20},{0,3,6,9,14,17,19,21}};

extern "C" void kernel_launch(void* const* d_in, const int* in_sizes, int n_in,
                              void* d_out, int out_size, void* d_ws, size_t ws_size,
                              hipStream_t stream) {
  (void)in_sizes; (void)n_in; (void)out_size; (void)ws_size;
  const float* pose    = (const float*)d_in[0];
  const float* sel     = (const float*)d_in[1];
  const float* elapsed = (const float*)d_in[2];
  const float* horizon = (const float*)d_in[3];
  const float* nmean   = (const float*)d_in[4];
  const float* nvar    = (const float*)d_in[5];
  const float* Wl      = (const float*)d_in[6];
  const float* Ul      = (const float*)d_in[7];
  const float* bl      = (const float*)d_in[8];

  char* ws = (char*)d_ws;
  auto alloc = [&](size_t bytes) {
    char* p = ws; ws += (bytes + 255) & ~(size_t)255; return p;
  };
  h16*   xpad  = (h16*)alloc((size_t)MM * KX * 2);
  h16*   wp    = (h16*)alloc((size_t)KX * N4 * 2);      // swizzled
  h16*   u16   = (h16*)alloc((size_t)HH * N4 * 2);      // plain
  h16*   xw    = (h16*)alloc((size_t)MM * N4 * 2);
  h16*   hbuf  = (h16*)alloc((size_t)2 * BB * HH * 2);
  h16*   feat  = (h16*)alloc((size_t)MM * FD * 2);
  h16*   w0p   = (h16*)alloc((size_t)24 * FD * 64 * 2); // swizzled per joint
  h16*   w1p   = (h16*)alloc((size_t)24 * 64 * 32 * 2); // swizzled per joint
  h16*   w2p   = (h16*)alloc((size_t)24 * 32 * 32 * 2); // swizzled per joint
  float* deltas = (float*)alloc((size_t)MM * 72 * 4);
  unsigned int* cnt = (unsigned int*)alloc(256);
  hipMemsetAsync(cnt, 0, 256, stream);

  prep_x<<<(MM * KX + 255) / 256, 256, 0, stream>>>(pose, sel, elapsed, nmean, nvar, xpad);
  f32_to_f16_swz<<<(KX * N4 + 255) / 256, 256, 0, stream>>>(Wl, wp, 73, N4, KX);
  f32_to_f16_pad<<<(HH * N4 + 255) / 256, 256, 0, stream>>>(Ul, u16, HH, N4, HH);
  feat_base<<<(MM + 255) / 256, 256, 0, stream>>>(elapsed, horizon, feat);
  for (int j = 0; j < 24; ++j) {
    int din = 2 + HH + 3 * PN[j];
    f32_to_f16_swz<<<(FD * 64 + 255) / 256, 256, 0, stream>>>(
        (const float*)d_in[9 + 8 * j + 0], w0p + (size_t)j * FD * 64, din, 64, FD);
    f32_to_f16_swz<<<(64 * 32 + 255) / 256, 256, 0, stream>>>(
        (const float*)d_in[9 + 8 * j + 2], w1p + (size_t)j * 64 * 32, 64, 32, 64);
    f32_to_f16_swz<<<(32 * 32 + 255) / 256, 256, 0, stream>>>(
        (const float*)d_in[9 + 8 * j + 4], w2p + (size_t)j * 32 * 32, 32, 32, 32);
  }

  xw_gemm<<<dim3(MM / 16, N4 / 64), 128, 0, stream>>>(xpad, wp, bl, xw);

  size_t lds = (size_t)HH * 128 * 2 + (size_t)BB * HH * 2
             + (size_t)BB * 128 * 4 + (size_t)BB * 32 * 4 + 128 * 4;  // ~306 KB < 320 KB
  lstm_kernel<<<32, 128, lds, stream>>>(xw, u16, bl, hbuf, feat, cnt);

  for (int j = 0; j < 24; ++j) {
    Parents par; par.np = PN[j];
    for (int k = 0; k < 8; ++k) par.p[k] = (k < PN[j]) ? PL[j][k] : 0;
    fill_parents<<<(MM + 255) / 256, 256, 0, stream>>>(deltas, feat, par);
    int kpad = ((2 + HH + 3 * PN[j]) + 31) / 32 * 32;
    joint_mlp<<<MM / 16, 32, 0, stream>>>(
        feat, w0p + (size_t)j * FD * 64,
        (const float*)d_in[9 + 8 * j + 1], w1p + (size_t)j * 64 * 32,
        (const float*)d_in[9 + 8 * j + 3], w2p + (size_t)j * 32 * 32,
        (const float*)d_in[9 + 8 * j + 5], (const float*)d_in[9 + 8 * j + 6],
        (const float*)d_in[9 + 8 * j + 7], sel, deltas, 3 * j, kpad);
  }
  final_pred<<<(MM * 72 + 255) / 256, 256, 0, stream>>>(pose, sel, deltas, (float*)d_out);
}
